// AgentAttention_36000415875532
// MI455X (gfx1250) — matrix operand order repaired and checked by reading, not compiled, not run on hardware
//
#include <hip/hip_runtime.h>
#include <hip/hip_bf16.h>

// ---------------------------------------------------------------------------
// Types for CDNA5 WMMA (gfx1250, wave32)
// ---------------------------------------------------------------------------
typedef __bf16 v16bf __attribute__((ext_vector_type(16)));
typedef float  v8f   __attribute__((ext_vector_type(8)));

union FragB16 { v16bf v; uint4 q[2]; };

__device__ __forceinline__ v8f wmma_bf(const FragB16& a, const FragB16& b, v8f c) {
  return __builtin_amdgcn_wmma_f32_16x16x32_bf16(false, a.v, false, b.v, (short)0, c, false, false);
}

// native hardware conversion path (RNE)
__device__ __forceinline__ unsigned short f2bf(float f) {
  union { __bf16 h; unsigned short u; } cv;
  cv.h = (__bf16)f;
  return cv.u;
}
__device__ __forceinline__ float bf2f(unsigned short u) {
  return __uint_as_float(((unsigned)u) << 16);
}

__device__ __forceinline__ float halfmax16(float t) {
  t = fmaxf(t, __shfl_xor(t, 1));
  t = fmaxf(t, __shfl_xor(t, 2));
  t = fmaxf(t, __shfl_xor(t, 4));
  t = fmaxf(t, __shfl_xor(t, 8));
  return t;
}
__device__ __forceinline__ float halfsum16(float t) {
  t += __shfl_xor(t, 1);
  t += __shfl_xor(t, 2);
  t += __shfl_xor(t, 4);
  t += __shfl_xor(t, 8);
  return t;
}

// async global->LDS copy of one 16B chunk (ASYNCcnt-tracked DMA path)
__device__ __forceinline__ void async_cp16(unsigned lds_off, const void* g) {
  asm volatile("global_load_async_to_lds_b128 %0, %1, off"
               :: "v"(lds_off), "v"(g) : "memory");
}
__device__ __forceinline__ void async_wait0() {
  asm volatile("s_wait_asynccnt 0" ::: "memory");
}

// Problem constants
#define BB   16
#define HH   56
#define WW   56
#define NN   3136      // 56*56
#define CC   384
#define NH   12
#define HD   32
#define AA   49
#define APAD 64
#define SCALE 0.17677669529663687f  // 1/sqrt(32)

// ---------------------------------------------------------------------------
// Elementwise f32 -> bf16 conversion (for x).  4 elements / thread.
// ---------------------------------------------------------------------------
__global__ __launch_bounds__(256)
void cvt_bf16_kernel(const float* __restrict__ in, unsigned short* __restrict__ out, int n) {
  int id = (blockIdx.x * 256 + threadIdx.x) * 4;
  if (id < n) {
    float4 f = *(const float4*)(in + id);
    ushort4 h;
    h.x = f2bf(f.x); h.y = f2bf(f.y); h.z = f2bf(f.z); h.w = f2bf(f.w);
    *(ushort4*)(out + id) = h;
  }
}

// ---------------------------------------------------------------------------
// Weight prep: WT[n*K + k] = bf16(W[k*N + n])  (transpose+convert, tiny)
// ---------------------------------------------------------------------------
__global__ __launch_bounds__(256)
void wtrans_kernel(const float* __restrict__ W, unsigned short* __restrict__ WT,
                   int K, int N) {
  int id = blockIdx.x * 256 + threadIdx.x;
  if (id < K * N) {
    int nn = id / K, kk = id % K;
    WT[id] = f2bf(W[(size_t)kk * N + nn]);
  }
}

// ---------------------------------------------------------------------------
// Kernel 1: bf16 GEMM, double-buffered async-LDS staging, 128x128x32 block
// tile, 8 waves (4x2) each computing 32x64 via 2x4 WMMA tiles (1.5 LDS
// b128 loads per WMMA).  A: bf16 row-major [M][K].  BT: bf16 [N][K].
// ---------------------------------------------------------------------------
__global__ __launch_bounds__(256)
void gemm_bf16_kernel(const unsigned short* __restrict__ A,
                      const unsigned short* __restrict__ BT,
                      const float* __restrict__ bias, float* __restrict__ Cf,
                      unsigned short* __restrict__ Cbf, int M, int N, int K) {
  __shared__ __align__(16) unsigned short As[2][128 * 40];
  __shared__ __align__(16) unsigned short Bs[2][128 * 40];

  const int tid  = threadIdx.x;
  const int wave = tid >> 5;
  const int lane = tid & 31;
  const int l16  = lane & 15;
  const int hf   = lane >> 4;
  const int wm   = wave & 3;   // M sub-block of 32
  const int wn   = wave >> 2;  // N sub-block of 64
  const int mBase = blockIdx.x * 128;
  const int nBase = blockIdx.y * 128;

  auto issue = [&](int buf, int kc) {
    // A tile: 128 rows x 32 bf16 = 512 x 16B chunks; 2 per thread
    // B tile: 128 rows x 32 bf16 = 512 x 16B chunks; 2 per thread
#pragma unroll
    for (int j = 0; j < 2; ++j) {
      int id  = tid + j * 256;
      int row = id >> 2;
      int c8  = (id & 3) * 8;
      unsigned ldsA = (unsigned)(size_t)(const void*)&As[buf][row * 40 + c8];
      async_cp16(ldsA, A + (size_t)(mBase + row) * K + kc + c8);
      unsigned ldsB = (unsigned)(size_t)(const void*)&Bs[buf][row * 40 + c8];
      async_cp16(ldsB, BT + (size_t)(nBase + row) * K + kc + c8);
    }
  };

  v8f acc[2][4];
#pragma unroll
  for (int mt = 0; mt < 2; ++mt)
#pragma unroll
    for (int nt = 0; nt < 4; ++nt)
#pragma unroll
      for (int i = 0; i < 8; ++i) acc[mt][nt][i] = 0.f;

  issue(0, 0);
  int buf = 0;
  for (int kc = 0; kc < K; kc += 32) {
    async_wait0();
    __syncthreads();
    if (kc + 32 < K) issue(buf ^ 1, kc + 32);

    FragB16 fa[2], fb[4];
#pragma unroll
    for (int mt = 0; mt < 2; ++mt) {
      const unsigned short* p = &As[buf][(wm * 32 + mt * 16 + l16) * 40 + hf * 8];
      fa[mt].q[0] = *(const uint4*)p;
      fa[mt].q[1] = *(const uint4*)(p + 16);
    }
#pragma unroll
    for (int nt = 0; nt < 4; ++nt) {
      const unsigned short* p = &Bs[buf][(wn * 64 + nt * 16 + l16) * 40 + hf * 8];
      fb[nt].q[0] = *(const uint4*)p;
      fb[nt].q[1] = *(const uint4*)(p + 16);
    }
#pragma unroll
    for (int mt = 0; mt < 2; ++mt)
#pragma unroll
      for (int nt = 0; nt < 4; ++nt)
        acc[mt][nt] = wmma_bf(fa[mt], fb[nt], acc[mt][nt]);
    buf ^= 1;
  }

#pragma unroll
  for (int mt = 0; mt < 2; ++mt)
#pragma unroll
    for (int nt = 0; nt < 4; ++nt)
#pragma unroll
      for (int i = 0; i < 8; ++i) {
        int row = mBase + wm * 32 + mt * 16 + hf * 8 + i;
        int col = nBase + wn * 64 + nt * 16 + l16;
        float v = acc[mt][nt][i];
        if (bias) v += bias[col];
        if (Cf)  Cf[(size_t)row * N + col] = v;
        if (Cbf) Cbf[(size_t)row * N + col] = f2bf(v);
      }
}

// ---------------------------------------------------------------------------
// Kernel 2: agent pooling from bf16 q. Scaled bf16 ath, padded to 64 rows.
// grid (16, 64), block 384.
// ---------------------------------------------------------------------------
__global__ __launch_bounds__(384)
void pool_kernel(const unsigned short* __restrict__ qbf,
                 unsigned short* __restrict__ athbf) {
  const int b = blockIdx.x, a = blockIdx.y, ch = threadIdx.x;
  float v = 0.f;
  if (a < AA) {
    const int ar = a / 7, ac = a % 7;
    const int p0 = (ar * 8) * WW + ac * 8;
    float s = 0.f;
#pragma unroll
    for (int dy = 0; dy < 8; ++dy)
#pragma unroll
      for (int dx = 0; dx < 8; ++dx)
        s += bf2f(qbf[((size_t)b * NN + p0 + dy * WW + dx) * CC + ch]);
    v = s * (1.f / 64.f) * SCALE;
  }
  const int h = ch >> 5, k = ch & 31;
  athbf[(((size_t)b * NH + h) * APAD + a) * HD + k] = f2bf(v);
}

// ---------------------------------------------------------------------------
// Kernel 3: per-head transpose of bf16 V slice of kv -> vtbf[b,h,d,n].
// grid (192, 49), block 256.
// ---------------------------------------------------------------------------
__global__ __launch_bounds__(256)
void vtrans_kernel(const unsigned short* __restrict__ kvbf,
                   unsigned short* __restrict__ vtbf) {
  __shared__ unsigned short T[32 * 65];
  const int bh = blockIdx.x;
  const int b = bh / NH, h = bh % NH;
  const int p0 = blockIdx.y * 64;
  const int tx = threadIdx.x & 31;   // channel d
  const int ty = threadIdx.x >> 5;   // 0..7
  for (int pp = ty; pp < 64; pp += 8)
    T[tx * 65 + pp] = kvbf[((size_t)b * NN + p0 + pp) * (2 * CC) + CC + h * HD + tx];
  __syncthreads();
  for (int idx = threadIdx.x; idx < 2048; idx += 256) {
    int d = idx >> 6, pp = idx & 63;
    vtbf[((size_t)bh * HD + d) * NN + p0 + pp] = T[d * 65 + pp];
  }
}

// ---------------------------------------------------------------------------
// Kernel 4: bias pipelines (inner 7x7->7x7 resize is identity).
// posb[h,a,p] = bilerp7to56(an[h,a])(p) + ah[h,a,pr] + aw[h,a,pc]
// agb [h,p,a] = bilerp7to56(na[h,a])(p) + ha[h,pr,a] + wa[h,pc,a]
// grid 588, block 256.
// ---------------------------------------------------------------------------
__device__ __forceinline__ float bilerp7(const float* t, int r0, int r1, float wr,
                                         int c0, int c1, float wc) {
  float v0 = t[r0 * 7 + c0] * (1.f - wc) + t[r0 * 7 + c1] * wc;
  float v1 = t[r1 * 7 + c0] * (1.f - wc) + t[r1 * 7 + c1] * wc;
  return v0 * (1.f - wr) + v1 * wr;
}

__global__ __launch_bounds__(256)
void bias_kernel(const float* __restrict__ an, const float* __restrict__ na,
                 const float* __restrict__ ahb, const float* __restrict__ awb,
                 const float* __restrict__ hab, const float* __restrict__ wab,
                 float* __restrict__ posb, float* __restrict__ agb) {
  const int ha = blockIdx.x;
  const int h = ha / AA, a = ha % AA;
  const float* ant = an + (size_t)ha * 49;
  const float* nat = na + (size_t)ha * 49;
  for (int p = threadIdx.x; p < NN; p += 256) {
    const int pr = p / WW, pc = p % WW;
    float sr = fmaxf((pr + 0.5f) * 0.125f - 0.5f, 0.f);
    float sc = fmaxf((pc + 0.5f) * 0.125f - 0.5f, 0.f);
    int r0 = min((int)sr, 6), r1 = min(r0 + 1, 6);
    int c0 = min((int)sc, 6), c1 = min(c0 + 1, 6);
    float wr = sr - (float)r0, wc = sc - (float)c0;
    float pv = bilerp7(ant, r0, r1, wr, c0, c1, wc)
             + ahb[(size_t)ha * HH + pr] + awb[(size_t)ha * WW + pc];
    posb[(size_t)ha * NN + p] = pv;
    float av = bilerp7(nat, r0, r1, wr, c0, c1, wc)
             + hab[((size_t)h * HH + pr) * AA + a] + wab[((size_t)h * WW + pc) * AA + a];
    agb[((size_t)h * NN + p) * AA + a] = av;
  }
}

// ---------------------------------------------------------------------------
// Kernel 5: flash-style agent attention per (b,h).
// S = ath(64x32,scaled) @ k^T(32xn) + pos_bias ; online softmax over n ;
// agent_v = P @ v(nx32).  8 waves split n in chunks of 32, merged in LDS.
// Output: avtbf[b,h,d,a] bf16 (transposed, padded to 64 agents).
// grid 192, block 256.
// ---------------------------------------------------------------------------
__global__ __launch_bounds__(256)
void agent_attn_kernel(const unsigned short* __restrict__ athbf,
                       const unsigned short* __restrict__ kvbf,
                       const unsigned short* __restrict__ vtbf,
                       const float* __restrict__ posb,
                       unsigned short* __restrict__ avtbf) {
  __shared__ __align__(16) unsigned char smem[69632];
  unsigned short* Pl = (unsigned short*)smem;       // 8*64*40 ushort = 40960 B
  float* Ob = (float*)smem;                          // 8*64*32 f32 = 65536 B (reuse)
  float* Mb = (float*)(smem + 65536);
  float* Lb = (float*)(smem + 65536 + 2048);

  const int bh = blockIdx.x;
  const int b = bh / NH, h = bh % NH;
  const int tid = threadIdx.x, wave = tid >> 5, lane = tid & 31;
  const int l16 = lane & 15, hf = lane >> 4;

  FragB16 fam[4];
#pragma unroll
  for (int mt = 0; mt < 4; ++mt) {
    const unsigned short* p = athbf + (((size_t)bh * APAD) + mt * 16 + l16) * HD + hf * 8;
    fam[mt].q[0] = *(const uint4*)p;
    fam[mt].q[1] = *(const uint4*)(p + 16);
  }

  v8f o[4][2];
  float mrow[4][8], lrow[4][8];
#pragma unroll
  for (int mt = 0; mt < 4; ++mt) {
#pragma unroll
    for (int i = 0; i < 8; ++i) { mrow[mt][i] = -3.0e38f; lrow[mt][i] = 0.f; }
#pragma unroll
    for (int nt = 0; nt < 2; ++nt)
#pragma unroll
      for (int i = 0; i < 8; ++i) o[mt][nt][i] = 0.f;
  }

  unsigned short* Pw = Pl + wave * (64 * 40);

  for (int n0 = wave * 32; n0 < NN; n0 += 256) {
    FragB16 fkb[2];
#pragma unroll
    for (int nt = 0; nt < 2; ++nt) {
      int ncol = n0 + nt * 16 + l16;
      const unsigned short* p = kvbf + ((size_t)b * NN + ncol) * (2 * CC) + h * HD + hf * 8;
      fkb[nt].q[0] = *(const uint4*)p;
      fkb[nt].q[1] = *(const uint4*)(p + 16);
    }
    v8f s[4][2];
#pragma unroll
    for (int mt = 0; mt < 4; ++mt)
#pragma unroll
      for (int nt = 0; nt < 2; ++nt) {
        v8f z;
#pragma unroll
        for (int i = 0; i < 8; ++i) z[i] = 0.f;
        s[mt][nt] = wmma_bf(fam[mt], fkb[nt], z);
      }
#pragma unroll
    for (int mt = 0; mt < 4; ++mt)
#pragma unroll
      for (int i = 0; i < 8; ++i) {
        int a = mt * 16 + hf * 8 + i;
        if (a < AA) {
          const float* pb = posb + ((size_t)h * AA + a) * NN + n0 + l16;
          s[mt][0][i] += pb[0];
          s[mt][1][i] += pb[16];
        }
      }
#pragma unroll
    for (int mt = 0; mt < 4; ++mt)
#pragma unroll
      for (int i = 0; i < 8; ++i) {
        float t = fmaxf(s[mt][0][i], s[mt][1][i]);
        t = halfmax16(t);
        float mn = fmaxf(mrow[mt][i], t);
        float al = __expf(mrow[mt][i] - mn);
        mrow[mt][i] = mn;
        o[mt][0][i] *= al;
        o[mt][1][i] *= al;
        float p0 = __expf(s[mt][0][i] - mn);
        float p1 = __expf(s[mt][1][i] - mn);
        s[mt][0][i] = p0; s[mt][1][i] = p1;
        float u = halfsum16(p0 + p1);
        lrow[mt][i] = lrow[mt][i] * al + u;
      }
#pragma unroll
    for (int mt = 0; mt < 4; ++mt)
#pragma unroll
      for (int nt = 0; nt < 2; ++nt)
#pragma unroll
        for (int i = 0; i < 8; ++i)
          Pw[(mt * 16 + hf * 8 + i) * 40 + nt * 16 + l16] = f2bf(s[mt][nt][i]);

    FragB16 fpa[4], fvb[2];
#pragma unroll
    for (int mt = 0; mt < 4; ++mt) {
      const unsigned short* p = Pw + (mt * 16 + l16) * 40 + hf * 8;
      fpa[mt].q[0] = *(const uint4*)p;
      fpa[mt].q[1] = *(const uint4*)(p + 16);
    }
#pragma unroll
    for (int nt = 0; nt < 2; ++nt) {
      int d = nt * 16 + l16;
      const unsigned short* p = vtbf + ((size_t)bh * HD + d) * NN + n0 + hf * 8;
      fvb[nt].q[0] = *(const uint4*)p;
      fvb[nt].q[1] = *(const uint4*)(p + 16);
    }
#pragma unroll
    for (int mt = 0; mt < 4; ++mt)
#pragma unroll
      for (int nt = 0; nt < 2; ++nt)
        o[mt][nt] = wmma_bf(fpa[mt], fvb[nt], o[mt][nt]);
  }

  __syncthreads();
#pragma unroll
  for (int mt = 0; mt < 4; ++mt) {
#pragma unroll
    for (int nt = 0; nt < 2; ++nt)
#pragma unroll
      for (int i = 0; i < 8; ++i) {
        int row = mt * 16 + hf * 8 + i;
        Ob[((size_t)wave * 64 + row) * 32 + nt * 16 + l16] = o[mt][nt][i];
      }
    if (l16 == 0)
#pragma unroll
      for (int i = 0; i < 8; ++i) {
        int row = mt * 16 + hf * 8 + i;
        Mb[wave * 64 + row] = mrow[mt][i];
        Lb[wave * 64 + row] = lrow[mt][i];
      }
  }
  __syncthreads();
  for (int idx = tid; idx < 2048; idx += 256) {
    int row = idx >> 5, d = idx & 31;
    float gm = -3.0e38f;
#pragma unroll
    for (int w = 0; w < 8; ++w) gm = fmaxf(gm, Mb[w * 64 + row]);
    float L = 0.f, O = 0.f;
#pragma unroll
    for (int w = 0; w < 8; ++w) {
      float e = __expf(Mb[w * 64 + row] - gm);
      L += Lb[w * 64 + row] * e;
      O += Ob[((size_t)w * 64 + row) * 32 + d] * e;
    }
    avtbf[((size_t)bh * HD + d) * APAD + row] = f2bf(O / L);
  }
}

// ---------------------------------------------------------------------------
// Kernel 6: q attention. One 16-row n-tile per wave.  grid 4704, block 256.
// ---------------------------------------------------------------------------
__global__ __launch_bounds__(256)
void qattn_kernel(const unsigned short* __restrict__ qbf,
                  const unsigned short* __restrict__ athbf,
                  const unsigned short* __restrict__ avtbf,
                  const float* __restrict__ agb,
                  float* __restrict__ obuf) {
  __shared__ __align__(16) unsigned short Pl[8 * 16 * 40];
  const int wave = threadIdx.x >> 5, lane = threadIdx.x & 31;
  const int l16 = lane & 15, hf = lane >> 4;
  const int gw = blockIdx.x * 8 + wave;
  const int bh = gw / 196, tile = gw % 196;
  const int b = bh / NH, h = bh % NH;
  const int n0 = tile * 16;

  FragB16 fqa;
  {
    const unsigned short* p = qbf + ((size_t)b * NN + n0 + l16) * CC + h * HD + hf * 8;
    fqa.q[0] = *(const uint4*)p;
    fqa.q[1] = *(const uint4*)(p + 16);
  }
  v8f s[4];
#pragma unroll
  for (int nt = 0; nt < 4; ++nt) {
    FragB16 fb;
    const unsigned short* p = athbf + (((size_t)bh * APAD) + nt * 16 + l16) * HD + hf * 8;
    fb.q[0] = *(const uint4*)p;
    fb.q[1] = *(const uint4*)(p + 16);
    v8f z;
#pragma unroll
    for (int i = 0; i < 8; ++i) z[i] = 0.f;
    s[nt] = wmma_bf(fqa, fb, z);
  }
#pragma unroll
  for (int nt = 0; nt < 4; ++nt) {
    int a = nt * 16 + l16;
#pragma unroll
    for (int i = 0; i < 8; ++i) {
      int nrow = n0 + hf * 8 + i;
      if (a < AA) s[nt][i] += agb[((size_t)h * NN + nrow) * AA + a];
      else        s[nt][i] = -1.0e30f;
    }
  }
  unsigned short* Pw = Pl + wave * (16 * 40);
#pragma unroll
  for (int i = 0; i < 8; ++i) {
    float t = fmaxf(fmaxf(s[0][i], s[1][i]), fmaxf(s[2][i], s[3][i]));
    t = halfmax16(t);
    float p0 = __expf(s[0][i] - t), p1 = __expf(s[1][i] - t);
    float p2 = __expf(s[2][i] - t), p3 = __expf(s[3][i] - t);
    float l = halfsum16(p0 + p1 + p2 + p3);
    float inv = 1.0f / l;
    s[0][i] = p0 * inv; s[1][i] = p1 * inv; s[2][i] = p2 * inv; s[3][i] = p3 * inv;
  }
#pragma unroll
  for (int nt = 0; nt < 4; ++nt)
#pragma unroll
    for (int i = 0; i < 8; ++i)
      Pw[(hf * 8 + i) * 40 + nt * 16 + l16] = f2bf(s[nt][i]);

  v8f oacc[2];
#pragma unroll
  for (int nt = 0; nt < 2; ++nt)
#pragma unroll
    for (int i = 0; i < 8; ++i) oacc[nt][i] = 0.f;
#pragma unroll
  for (int kc = 0; kc < 2; ++kc) {
    FragB16 fpa;
    const unsigned short* pp = Pw + l16 * 40 + kc * 32 + hf * 8;
    fpa.q[0] = *(const uint4*)pp;
    fpa.q[1] = *(const uint4*)(pp + 16);
#pragma unroll
    for (int nt = 0; nt < 2; ++nt) {
      FragB16 fvb;
      int d = nt * 16 + l16;
      const unsigned short* p = avtbf + ((size_t)bh * HD + d) * APAD + kc * 32 + hf * 8;
      fvb.q[0] = *(const uint4*)p;
      fvb.q[1] = *(const uint4*)(p + 16);
      oacc[nt] = wmma_bf(fpa, fvb, oacc[nt]);
    }
  }
#pragma unroll
  for (int nt = 0; nt < 2; ++nt)
#pragma unroll
    for (int i = 0; i < 8; ++i) {
      int nrow = n0 + hf * 8 + i;
      obuf[((size_t)b * NN + nrow) * CC + h * HD + nt * 16 + l16] = oacc[nt][i];
    }
}

// ---------------------------------------------------------------------------
// Kernel 7: depthwise 3x3 conv on bf16 v, add to qattn output, emit bf16 A
// matrix for the projection GEMM.  grid b*H = 896, block 384.
// ---------------------------------------------------------------------------
__global__ __launch_bounds__(384)
void dwc_kernel(const unsigned short* __restrict__ kvbf,
                const float* __restrict__ dwc_w, const float* __restrict__ dwc_b,
                const float* __restrict__ obuf, unsigned short* __restrict__ obf) {
  const int by = blockIdx.x;
  const int b = by / HH, y = by % HH;
  const int ch = threadIdx.x;
  float w[9];
#pragma unroll
  for (int j = 0; j < 9; ++j) w[j] = dwc_w[ch * 9 + j];
  const float bias = dwc_b[ch];
  for (int x = 0; x < WW; ++x) {
    float acc = bias;
#pragma unroll
    for (int dy = -1; dy <= 1; ++dy) {
      int yy = y + dy;
      if (yy < 0 || yy >= HH) continue;
#pragma unroll
      for (int dx = -1; dx <= 1; ++dx) {
        int xx = x + dx;
        if (xx < 0 || xx >= WW) continue;
        acc += w[(dy + 1) * 3 + (dx + 1)] *
               bf2f(kvbf[((size_t)b * NN + yy * WW + xx) * (2 * CC) + CC + ch]);
      }
    }
    size_t idx = ((size_t)b * NN + y * WW + x) * CC + ch;
    obf[idx] = f2bf(obuf[idx] + acc);
  }
}

// ---------------------------------------------------------------------------
// Launch
// ---------------------------------------------------------------------------
extern "C" void kernel_launch(void* const* d_in, const int* in_sizes, int n_in,
                              void* d_out, int out_size, void* d_ws, size_t ws_size,
                              hipStream_t stream) {
  (void)in_sizes; (void)n_in; (void)out_size; (void)ws_size;
  const float* x     = (const float*)d_in[0];
  const float* Wq    = (const float*)d_in[1];
  const float* Wkv   = (const float*)d_in[2];
  const float* Wproj = (const float*)d_in[3];
  const float* bproj = (const float*)d_in[4];
  const float* dwcw  = (const float*)d_in[5];
  const float* dwcb  = (const float*)d_in[6];
  const float* an    = (const float*)d_in[7];
  const float* na    = (const float*)d_in[8];
  const float* ahb   = (const float*)d_in[9];
  const float* awb   = (const float*)d_in[10];
  const float* hab   = (const float*)d_in[11];
  const float* wab   = (const float*)d_in[12];
  float* out = (float*)d_out;

  char* ws = (char*)d_ws;
  const size_t EL   = (size_t)BB * NN * CC;      // 19,267,584
  size_t off = 0;
  unsigned short* xbf    = (unsigned short*)(ws + off); off += EL * 2;
  unsigned short* qbf    = (unsigned short*)(ws + off); off += EL * 2;
  unsigned short* kvbf   = (unsigned short*)(ws + off); off += EL * 4;
  unsigned short* vtbf   = (unsigned short*)(ws + off); off += EL * 2;
  unsigned short* athbf  = (unsigned short*)(ws + off); off += (size_t)BB * NH * APAD * HD * 2;
  float*          posb   = (float*)(ws + off);          off += (size_t)NH * AA * NN * 4;
  float*          agb    = (float*)(ws + off);          off += (size_t)NH * AA * NN * 4;
  unsigned short* avtbf  = (unsigned short*)(ws + off); off += (size_t)BB * NH * HD * APAD * 2;
  float*          obuf   = (float*)(ws + off);          off += EL * 4;
  unsigned short* obf    = (unsigned short*)(ws + off); off += EL * 2;
  unsigned short* WqT    = (unsigned short*)(ws + off); off += (size_t)CC * CC * 2;
  unsigned short* WkvT   = (unsigned short*)(ws + off); off += (size_t)CC * 2 * CC * 2;
  unsigned short* WprojT = (unsigned short*)(ws + off); off += (size_t)CC * CC * 2;

  const int M = BB * NN;  // 50176
  cvt_bf16_kernel<<<(int)(EL / 1024), 256, 0, stream>>>(x, xbf, (int)EL);
  wtrans_kernel<<<(CC * CC + 255) / 256, 256, 0, stream>>>(Wq, WqT, CC, CC);
  wtrans_kernel<<<(CC * 2 * CC + 255) / 256, 256, 0, stream>>>(Wkv, WkvT, CC, 2 * CC);
  wtrans_kernel<<<(CC * CC + 255) / 256, 256, 0, stream>>>(Wproj, WprojT, CC, CC);

  gemm_bf16_kernel<<<dim3(M / 128, CC / 128), 256, 0, stream>>>(
      xbf, WqT, nullptr, nullptr, qbf, M, CC, CC);
  gemm_bf16_kernel<<<dim3(M / 128, 2 * CC / 128), 256, 0, stream>>>(
      xbf, WkvT, nullptr, nullptr, kvbf, M, 2 * CC, CC);

  pool_kernel<<<dim3(BB, APAD), 384, 0, stream>>>(qbf, athbf);
  vtrans_kernel<<<dim3(BB * NH, NN / 64), 256, 0, stream>>>(kvbf, vtbf);
  bias_kernel<<<NH * AA, 256, 0, stream>>>(an, na, ahb, awb, hab, wab, posb, agb);
  agent_attn_kernel<<<BB * NH, 256, 0, stream>>>(athbf, kvbf, vtbf, posb, avtbf);
  qattn_kernel<<<(BB * NH * (NN / 16)) / 8, 256, 0, stream>>>(qbf, athbf, avtbf, agb, obuf);
  dwc_kernel<<<BB * HH, 384, 0, stream>>>(kvbf, dwcw, dwcb, obuf, obf);

  gemm_bf16_kernel<<<dim3(M / 128, CC / 128), 256, 0, stream>>>(
      obf, WprojT, bproj, out, nullptr, M, CC, CC);
}
